// KANGraphAttentionLayerARC1_54717883351547
// MI455X (gfx1250) — matrix operand order repaired and checked by reading, not compiled, not run on hardware
//
#include <hip/hip_runtime.h>
#include <hip/hip_bf16.h>

#define N_NODES 10000
#define IN_F 512
#define OUT_F 512
#define N_EDGES 160000
#define N_COEF 8               // GRID_SIZE + SPLINE_ORDER
#define KFLAT (IN_F * N_COEF)  // 4096
#define ALPHA 0.2f

typedef __attribute__((ext_vector_type(16))) __bf16 v16bf;
typedef __attribute__((ext_vector_type(8)))  __bf16 v8bf;
typedef __attribute__((ext_vector_type(8)))  float  v8f;

// ---------------------------------------------------------------------------
// Utility kernels
// ---------------------------------------------------------------------------
__global__ void k_f32_to_bf16(const float* __restrict__ src, __bf16* __restrict__ dst, size_t n) {
    size_t i = (size_t)blockIdx.x * blockDim.x + threadIdx.x;
    size_t stride = (size_t)gridDim.x * blockDim.x;
    for (; i < n; i += stride) dst[i] = (__bf16)src[i];
}

__global__ void k_zero_u32(unsigned* __restrict__ p, size_t n) {
    size_t i = (size_t)blockIdx.x * blockDim.x + threadIdx.x;
    size_t stride = (size_t)gridDim.x * blockDim.x;
    for (; i < n; i += stride) p[i] = 0u;
}

// Transpose f32 [R,C] -> bf16 [C,R] via LDS tile (padded to dodge bank conflicts)
__global__ void k_transpose_f32_to_bf16(const float* __restrict__ src, __bf16* __restrict__ dst,
                                        int R, int C) {
    __shared__ float tile[32][33];
    int bx = blockIdx.x * 32;   // C block
    int by = blockIdx.y * 32;   // R block
    int tx = threadIdx.x;       // 0..31
    for (int i = threadIdx.y; i < 32; i += blockDim.y)
        tile[i][tx] = src[(size_t)(by + i) * C + bx + tx];
    __syncthreads();
    for (int i = threadIdx.y; i < 32; i += blockDim.y)
        dst[(size_t)(bx + i) * R + by + tx] = (__bf16)tile[tx][i];
}

// ---------------------------------------------------------------------------
// WMMA bf16 NT-GEMM:  C[M,N](f32) = A[M,K](bf16,row) * Bt[N,K](bf16,row)^T (+Cin)
// One wave computes a 16x64 output tile: one shared A-frag, 4 B-frags,
// 4 independent accumulators -> 4x less A re-read from L2 and 4-deep
// independent WMMA chains per K-step. M % 16 == 0, N % 64 == 0, K % 32 == 0.
// ---------------------------------------------------------------------------
__global__ __launch_bounds__(32)
void k_gemm_bf16_nt(const __bf16* __restrict__ A, const __bf16* __restrict__ Bt,
                    const float* __restrict__ Cin, float* __restrict__ Cout,
                    int M, int N, int K) {
    const int tn   = blockIdx.x * 64;
    const int tm   = blockIdx.y * 16;
    const int lane = threadIdx.x;     // 0..31, wave32
    const int half = lane >> 4;       // 0 | 1
    const int l    = lane & 15;

    // A-frag lane layout: contiguous K{k0+16h..+15}
    const __bf16* pa = A + (size_t)(tm + l) * K + 16 * half;
    // B-frag lane layout: contiguous K{k0+16h..+15}, column tn + 16q + l
    const __bf16* pb = Bt + (size_t)(tn + l) * K + 16 * half;
    const size_t bcol = (size_t)16 * K;   // stride between 16-col subtiles

    union { v16bf v; v8bf h[2]; } af, bf0, bf1, bf2, bf3;
    v8f c0, c1, c2, c3;
    if (Cin) {
        const float* ci = Cin + (size_t)(tm + 8 * half) * N + tn + l;
        #pragma unroll
        for (int r = 0; r < 8; ++r) {
            c0[r] = ci[(size_t)r * N];
            c1[r] = ci[(size_t)r * N + 16];
            c2[r] = ci[(size_t)r * N + 32];
            c3[r] = ci[(size_t)r * N + 48];
        }
    } else {
        c0 = (v8f){0.f,0.f,0.f,0.f,0.f,0.f,0.f,0.f};
        c1 = c0; c2 = c0; c3 = c0;
    }

    for (int k0 = 0; k0 < K; k0 += 32) {
        af.h[0]  = *(const v8bf*)(pa);
        af.h[1]  = *(const v8bf*)(pa + 8);
        bf0.h[0] = *(const v8bf*)(pb);
        bf0.h[1] = *(const v8bf*)(pb + 8);
        bf1.h[0] = *(const v8bf*)(pb + bcol);
        bf1.h[1] = *(const v8bf*)(pb + bcol + 8);
        bf2.h[0] = *(const v8bf*)(pb + 2 * bcol);
        bf2.h[1] = *(const v8bf*)(pb + 2 * bcol + 8);
        bf3.h[0] = *(const v8bf*)(pb + 3 * bcol);
        bf3.h[1] = *(const v8bf*)(pb + 3 * bcol + 8);
        __builtin_prefetch(pa + 32, 0, 3);
        __builtin_prefetch(pb + 32, 0, 3);
        c0 = __builtin_amdgcn_wmma_f32_16x16x32_bf16(false, af.v, false, bf0.v, (short)0, c0, false, false);
        c1 = __builtin_amdgcn_wmma_f32_16x16x32_bf16(false, af.v, false, bf1.v, (short)0, c1, false, false);
        c2 = __builtin_amdgcn_wmma_f32_16x16x32_bf16(false, af.v, false, bf2.v, (short)0, c2, false, false);
        c3 = __builtin_amdgcn_wmma_f32_16x16x32_bf16(false, af.v, false, bf3.v, (short)0, c3, false, false);
        pa += 32;
        pb += 32;
    }

    float* co = Cout + (size_t)(tm + 8 * half) * N + tn + l;
    #pragma unroll
    for (int r = 0; r < 8; ++r) {
        co[(size_t)r * N]      = c0[r];
        co[(size_t)r * N + 16] = c1[r];
        co[(size_t)r * N + 32] = c2[r];
        co[(size_t)r * N + 48] = c3[r];
    }
}

// ---------------------------------------------------------------------------
// Attention logits per node: s = HW . a[:OUT], t = HW . a[OUT:]
// ---------------------------------------------------------------------------
__global__ __launch_bounds__(32)
void k_attn_logits(const float* __restrict__ HW, const float* __restrict__ a,
                   float* __restrict__ s, float* __restrict__ t) {
    int node = blockIdx.x;
    int lane = threadIdx.x;
    const float* row = HW + (size_t)node * OUT_F;
    float ss = 0.f, tt = 0.f;
    for (int i = lane; i < OUT_F; i += 32) {
        float v = row[i];
        ss += v * a[i];
        tt += v * a[OUT_F + i];
    }
    #pragma unroll
    for (int off = 16; off > 0; off >>= 1) {
        ss += __shfl_down(ss, off, 32);
        tt += __shfl_down(tt, off, 32);
    }
    if (lane == 0) { s[node] = ss; t[node] = tt; }
}

// order-preserving float<->uint keys for atomic max over signed floats
__device__ __forceinline__ unsigned fkey_enc(float f) {
    unsigned u = __float_as_uint(f);
    return (u & 0x80000000u) ? ~u : (u | 0x80000000u);
}
__device__ __forceinline__ float fkey_dec(unsigned k) {
    unsigned u = (k & 0x80000000u) ? (k ^ 0x80000000u) : ~k;
    return __uint_as_float(u);
}

// Edge pass 1: e = leaky_relu(s[row]+t[col]); segment max via atomic on keys
__global__ void k_edge_logits(const long long* __restrict__ ei,
                              const float* __restrict__ s, const float* __restrict__ t,
                              float* __restrict__ e, unsigned* __restrict__ mkey) {
    int idx = blockIdx.x * blockDim.x + threadIdx.x;
    if (idx >= N_EDGES) return;
    int r = (int)ei[idx];
    int c = (int)ei[N_EDGES + idx];
    float v = s[r] + t[c];
    v = (v > 0.f) ? v : ALPHA * v;
    e[idx] = v;
    atomicMax(&mkey[r], fkey_enc(v));
}

// Edge pass 2: ex = exp(e - m[row]); den[row] += ex
__global__ void k_edge_exp(const long long* __restrict__ ei,
                           float* __restrict__ e, const unsigned* __restrict__ mkey,
                           float* __restrict__ den) {
    int idx = blockIdx.x * blockDim.x + threadIdx.x;
    if (idx >= N_EDGES) return;
    int r = (int)ei[idx];
    float ex = __expf(e[idx] - fkey_dec(mkey[r]));
    e[idx] = ex;
    atomicAdd(&den[r], ex);
}

// Edge pass 3: agg[row,:] += (ex/den[row]) * h[col,:]   (128 thr/edge, float4)
__global__ __launch_bounds__(128)
void k_edge_aggregate(const long long* __restrict__ ei, const float* __restrict__ ex,
                      const float* __restrict__ den, const float* __restrict__ h,
                      float* __restrict__ agg) {
    int eidx = blockIdx.x;
    int r = (int)ei[eidx];
    int c = (int)ei[N_EDGES + eidx];
    float att = ex[eidx] / den[r];
    const float4* hv = (const float4*)(h + (size_t)c * IN_F);
    float* ag = agg + (size_t)r * IN_F + threadIdx.x * 4;
    float4 v = hv[threadIdx.x];
    atomicAdd(ag + 0, att * v.x);
    atomicAdd(ag + 1, att * v.y);
    atomicAdd(ag + 2, att * v.z);
    atomicAdd(ag + 3, att * v.w);
}

// ---------------------------------------------------------------------------
// KAN prep: sb = bf16(silu(agg)), basis[n, i*8+p] = bf16(B-spline_p(agg[n,i]))
// Cubic B-splines (Cox-de Boor), extended grid t_j = -1 + (j-3)*0.4, j=0..11
// ---------------------------------------------------------------------------
__global__ __launch_bounds__(256)
void k_kan_prep(const float* __restrict__ agg, __bf16* __restrict__ sb,
                __bf16* __restrict__ basis) {
    int n = blockIdx.x;
    for (int i = threadIdx.x; i < IN_F; i += 256) {
        float x = agg[(size_t)n * IN_F + i];
        float sig = 1.f / (1.f + __expf(-x));
        sb[(size_t)n * IN_F + i] = (__bf16)(x * sig);

        const float step = 0.4f;
        float tkn[12];
        #pragma unroll
        for (int j = 0; j < 12; ++j) tkn[j] = -1.f + (j - 3) * step;
        float b[11];
        #pragma unroll
        for (int j = 0; j < 11; ++j)
            b[j] = (x >= tkn[j] && x < tkn[j + 1]) ? 1.f : 0.f;
        #pragma unroll
        for (int p = 1; p <= 3; ++p) {
            #pragma unroll 10
            for (int j = 0; j < 11 - p; ++j) {
                float left  = (x - tkn[j]) / (tkn[j + p] - tkn[j]) * b[j];
                float right = (tkn[j + p + 1] - x) / (tkn[j + p + 1] - tkn[j + 1]) * b[j + 1];
                b[j] = left + right;
            }
        }
        __bf16* bo = basis + (size_t)n * KFLAT + (size_t)i * N_COEF;
        #pragma unroll
        for (int q = 0; q < N_COEF; ++q) bo[q] = (__bf16)b[q];
    }
}

// ---------------------------------------------------------------------------
// Host side
// ---------------------------------------------------------------------------
extern "C" void kernel_launch(void* const* d_in, const int* in_sizes, int n_in,
                              void* d_out, int out_size, void* d_ws, size_t ws_size,
                              hipStream_t stream) {
    const float*     h        = (const float*)d_in[0];
    const float*     W        = (const float*)d_in[1];
    const float*     a        = (const float*)d_in[2];
    const float*     base_w   = (const float*)d_in[3];
    const float*     spline_w = (const float*)d_in[4];
    const long long* eidx     = (const long long*)d_in[5];
    float* out = (float*)d_out;

    // workspace carve-out (256B aligned)
    char* w = (char*)d_ws;
    auto alloc = [&](size_t bytes) -> void* {
        void* p = (void*)w;
        w += (bytes + 255) & ~(size_t)255;
        return p;
    };
    __bf16* hbf    = (__bf16*)alloc((size_t)N_NODES * IN_F * 2);
    __bf16* Wtbf   = (__bf16*)alloc((size_t)OUT_F * IN_F * 2);
    float*  HW     = (float*) alloc((size_t)N_NODES * OUT_F * 4);
    float*  sbuf   = (float*) alloc((size_t)N_NODES * 4);
    float*  tbuf   = (float*) alloc((size_t)N_NODES * 4);
    float*  ebuf   = (float*) alloc((size_t)N_EDGES * 4);
    unsigned* mkey = (unsigned*)alloc((size_t)N_NODES * 4);
    float*  den    = (float*) alloc((size_t)N_NODES * 4);
    float*  agg    = (float*) alloc((size_t)N_NODES * IN_F * 4);
    __bf16* sb     = (__bf16*)alloc((size_t)N_NODES * IN_F * 2);
    __bf16* basis  = (__bf16*)alloc((size_t)N_NODES * KFLAT * 2);
    __bf16* bwbf   = (__bf16*)alloc((size_t)OUT_F * IN_F * 2);
    __bf16* swbf   = (__bf16*)alloc((size_t)OUT_F * KFLAT * 2);

    // 1) precision conversions for the WMMA path
    k_f32_to_bf16<<<2048, 256, 0, stream>>>(h, hbf, (size_t)N_NODES * IN_F);
    {
        dim3 g(IN_F / 32, OUT_F / 32), b(32, 8);
        // W is [IN,OUT]; we need Bt = W^T = [OUT,IN] (K-contiguous)
        k_transpose_f32_to_bf16<<<g, b, 0, stream>>>(W, Wtbf, IN_F, OUT_F);
    }
    k_f32_to_bf16<<<1024, 256, 0, stream>>>(base_w, bwbf, (size_t)OUT_F * IN_F);
    k_f32_to_bf16<<<2048, 256, 0, stream>>>(spline_w, swbf, (size_t)OUT_F * KFLAT);

    // 2) zero-init accumulators (ws is poisoned)
    k_zero_u32<<<2048, 256, 0, stream>>>((unsigned*)agg, (size_t)N_NODES * IN_F);
    k_zero_u32<<<64, 256, 0, stream>>>((unsigned*)den, N_NODES);
    k_zero_u32<<<64, 256, 0, stream>>>(mkey, N_NODES);

    // 3) HW = h @ W  (WMMA bf16, 16x64 per wave)
    {
        dim3 g(OUT_F / 64, N_NODES / 16);
        k_gemm_bf16_nt<<<g, 32, 0, stream>>>(hbf, Wtbf, nullptr, HW, N_NODES, OUT_F, IN_F);
    }

    // 4) attention logits
    k_attn_logits<<<N_NODES, 32, 0, stream>>>(HW, a, sbuf, tbuf);

    // 5) segment softmax over edges (3 passes)
    int eb = (N_EDGES + 255) / 256;
    k_edge_logits<<<eb, 256, 0, stream>>>(eidx, sbuf, tbuf, ebuf, mkey);
    k_edge_exp<<<eb, 256, 0, stream>>>(eidx, ebuf, mkey, den);
    k_edge_aggregate<<<N_EDGES, 128, 0, stream>>>(eidx, ebuf, den, h, agg);

    // 6) KAN activations: silu base path + B-spline basis (bf16)
    k_kan_prep<<<N_NODES, 256, 0, stream>>>(agg, sb, basis);

    // 7) out = silu(agg) @ base_w^T   (base_w already [OUT,IN] = Bt layout)
    {
        dim3 g(OUT_F / 64, N_NODES / 16);
        k_gemm_bf16_nt<<<g, 32, 0, stream>>>(sb, bwbf, nullptr, out, N_NODES, OUT_F, IN_F);
        // 8) out += basis @ spline_w (spline_w already [OUT, IN*8] = Bt layout)
        k_gemm_bf16_nt<<<g, 32, 0, stream>>>(basis, swbf, out, out, N_NODES, OUT_F, KFLAT);
    }

    (void)in_sizes; (void)n_in; (void)out_size; (void)ws_size;
}
